// SelfAttention_21878563406136
// MI455X (gfx1250) — compile-verified, matrix operand order checked by reading, not run on hardware
//
#include <hip/hip_runtime.h>

typedef __attribute__((ext_vector_type(16))) __bf16        v16bf;
typedef __attribute__((ext_vector_type(8)))  __bf16        v8bf;
typedef __attribute__((ext_vector_type(8)))  float         v8f;
typedef __attribute__((ext_vector_type(4)))  unsigned int  u32x4;
typedef __attribute__((ext_vector_type(8)))  int           i32x8;
typedef __attribute__((ext_vector_type(4)))  int           i32x4;

#define B_  8
#define S_  1024
#define D_  1024
#define H_  16
#define DH_ 64
#define N_  1024   /* H_*DH_ */
#define KROW 72    /* padded LDS row stride (elements) for the TDM K tile */

__device__ __forceinline__ unsigned short f2bf(float f) {
  union { float f; unsigned u; } x; x.f = f;
  unsigned r = x.u + 0x7FFFu + ((x.u >> 16) & 1u);   // round-to-nearest-even
  return (unsigned short)(r >> 16);
}

__device__ __forceinline__ v16bf cat8(v8bf lo, v8bf hi) {
  return __builtin_shufflevector(lo, hi, 0,1,2,3,4,5,6,7,8,9,10,11,12,13,14,15);
}

__device__ __forceinline__ v8f wmma_bf16(v16bf a, v16bf b, v8f c) {
  // D = A(16x32 bf16) * B(32x16 bf16) + C(16x16 f32)
  return __builtin_amdgcn_wmma_f32_16x16x32_bf16(false, a, false, b, (short)0, c,
                                                 false, false);
}

__device__ __forceinline__ v16bf load_bfrag(const unsigned short* p) {
  return cat8(*(const v8bf*)p, *(const v8bf*)(p + 16));
}

__device__ __forceinline__ v16bf load_afrag_f32(const float* p) {
  float ab[16];
  *(float4*)(ab + 0)  = *(const float4*)(p);
  *(float4*)(ab + 4)  = *(const float4*)(p + 4);
  *(float4*)(ab + 8)  = *(const float4*)(p + 16);
  *(float4*)(ab + 12) = *(const float4*)(p + 20);
  v16bf a;
  #pragma unroll
  for (int j = 0; j < 16; ++j) a[j] = (__bf16)ab[j];
  return a;
}

// ---------------------------------------------------------------------------
// TDM: async-load a 2D bf16 tile [32 keys x 64 dh], row stride 64, into LDS
// with 8-element padding per row (LDS row stride 72 el) so later
// ds_load_b128 fragment reads are bank-conflict-free.
// D# per CDNA5 ISA sec 8.3/8.4. Tracked with TENSORcnt.
// ---------------------------------------------------------------------------
__device__ __forceinline__ void tdm_load_k(const unsigned short* gsrc,
                                           unsigned short* ldst) {
  const unsigned lds_off = (unsigned)(uintptr_t)ldst;       // flat[31:0] == LDS byte offset
  const unsigned long long ga = (unsigned long long)(uintptr_t)gsrc;
  u32x4 g0;
  g0[0] = 1u;                                   // count=1, user desc
  g0[1] = lds_off;                              // lds_addr
  g0[2] = (unsigned)ga;                         // global_addr[31:0]
  g0[3] = (unsigned)(ga >> 32) | 0x80000000u;   // global_addr[56:32] | type=2
  i32x8 g1;
  g1[0] = 0x07110000;    // data_size=2B | pad_enable | pad_interval=16x8B | pad_amount=4dw
  g1[1] = 64 << 16;      // tensor_dim0 = 64 (low 16 bits)
  g1[2] = 32 << 16;      // tensor_dim1 = 32
  g1[3] = 64 << 16;      // tile_dim0 = 64
  g1[4] = 32;            // tile_dim1 = 32, tile_dim2 = 0
  g1[5] = 64;            // tensor_dim0_stride = 64
  g1[6] = 0;
  g1[7] = 0;
  i32x4 z4 = {};
#if __clang_major__ >= 23
  i32x8 z8 = {};
  __builtin_amdgcn_tensor_load_to_lds(g0, g1, z4, z4, z8, 0);
#else
  __builtin_amdgcn_tensor_load_to_lds(g0, g1, z4, z4, 0);
#endif
}

// ---------------------------------------------------------------------------
// Kernel 1: transpose-convert weights fp32 [D,N] -> bf16 [N,D] (x3 matrices)
// LDS-tiled 32x32 transpose: coalesced fp32 reads AND coalesced bf16 writes.
// ---------------------------------------------------------------------------
__global__ __launch_bounds__(256) void wconv_kernel(
    const float* __restrict__ WQ, const float* __restrict__ WK,
    const float* __restrict__ WV, unsigned short* __restrict__ Wt) {
  __shared__ float tile[32][33];                       // +1 pad: no bank conflicts
  const int bid = blockIdx.x;                          // 3 * 32 * 32 blocks
  const int mat = bid >> 10;
  const int rem = bid & 1023;
  const int k0  = (rem >> 5) << 5;
  const int n0  = (rem & 31) << 5;
  const int tx = threadIdx.x & 31;
  const int ty = threadIdx.x >> 5;                     // 0..7
  const float* W = (mat == 0) ? WQ : (mat == 1) ? WK : WV;
  #pragma unroll
  for (int j = 0; j < 4; ++j)                          // read: contiguous in n
    tile[ty + 8 * j][tx] = W[(size_t)(k0 + ty + 8 * j) * N_ + n0 + tx];
  __syncthreads();
  unsigned short* o = Wt + (size_t)mat * (D_ * N_);
  #pragma unroll
  for (int j = 0; j < 4; ++j)                          // write: contiguous in k
    o[(size_t)(n0 + ty + 8 * j) * D_ + k0 + tx] = f2bf(tile[tx][ty + 8 * j]);
}

// ---------------------------------------------------------------------------
// Kernel 2: QKV projection GEMM, K loop unrolled 2x with alternating register
// sets (no ping-pong copies): loads for step k+1 land while step k's WMMAs run.
// ---------------------------------------------------------------------------
__global__ __launch_bounds__(256) void proj_kernel(
    const float* __restrict__ Q, const float* __restrict__ K,
    const float* __restrict__ V, const unsigned short* __restrict__ Wt,
    unsigned short* __restrict__ qp, unsigned short* __restrict__ kp,
    unsigned short* __restrict__ vT) {
  const int lane = threadIdx.x & 31;
  const int wave = threadIdx.x >> 5;
  const int tile = blockIdx.x * 8 + wave;              // 3 * 512 * 16 tiles
  const int mat  = tile / (512 * 16);
  const int rem  = tile % (512 * 16);
  const int m0   = (rem >> 4) << 4;                    // 16-row tile of B*S
  const int n0   = (rem & 15) << 6;                    // 64-col tile of H*Dh
  const int half = lane >> 4, lrow = lane & 15, koff = half * 8;

  const float* A = (mat == 0) ? Q : (mat == 1) ? K : V;
  const unsigned short* WtB = Wt + (size_t)mat * (D_ * N_);
  const float* arow = A + (size_t)(m0 + lrow) * D_ + koff;
  const unsigned short* wrow[4];
  #pragma unroll
  for (int nt = 0; nt < 4; ++nt)
    wrow[nt] = WtB + (size_t)(n0 + nt * 16 + lrow) * D_ + koff;

  v8f acc[4] = {};
  v16bf a0 = load_afrag_f32(arow);
  v16bf b0[4], b1[4];
  v16bf a1;
  #pragma unroll
  for (int nt = 0; nt < 4; ++nt) b0[nt] = load_bfrag(wrow[nt]);

  int kc = 0;
  for (; kc + 64 < D_; kc += 64) {
    a1 = load_afrag_f32(arow + kc + 32);
    #pragma unroll
    for (int nt = 0; nt < 4; ++nt) b1[nt] = load_bfrag(wrow[nt] + kc + 32);
    __builtin_prefetch(arow + kc + 96, 0, 3);
    #pragma unroll
    for (int nt = 0; nt < 4; ++nt) acc[nt] = wmma_bf16(a0, b0[nt], acc[nt]);
    a0 = load_afrag_f32(arow + kc + 64);
    #pragma unroll
    for (int nt = 0; nt < 4; ++nt) b0[nt] = load_bfrag(wrow[nt] + kc + 64);
    #pragma unroll
    for (int nt = 0; nt < 4; ++nt) acc[nt] = wmma_bf16(a1, b1[nt], acc[nt]);
  }
  // kc == D_-64 here: a0/b0 hold chunk D_-64; last chunk D_-32 still to load
  a1 = load_afrag_f32(arow + D_ - 32);
  #pragma unroll
  for (int nt = 0; nt < 4; ++nt) b1[nt] = load_bfrag(wrow[nt] + D_ - 32);
  #pragma unroll
  for (int nt = 0; nt < 4; ++nt) acc[nt] = wmma_bf16(a0, b0[nt], acc[nt]);
  #pragma unroll
  for (int nt = 0; nt < 4; ++nt) acc[nt] = wmma_bf16(a1, b1[nt], acc[nt]);

  #pragma unroll
  for (int nt = 0; nt < 4; ++nt) {
    #pragma unroll
    for (int i = 0; i < 8; ++i) {
      const int m = m0 + i + half * 8;                 // C layout: row<->VGPR
      const int n = n0 + nt * 16 + lrow;               //           col<->lane
      const int b = m >> 10, s = m & 1023;
      const int h = n >> 6,  dh = n & 63;
      const unsigned short bv = f2bf(acc[nt][i]);
      if (mat == 0)      qp[(((size_t)(b * H_ + h) * S_ + s) << 6) + dh] = bv;
      else if (mat == 1) kp[(((size_t)(b * H_ + h) * S_ + s) << 6) + dh] = bv;
      else               vT[(((size_t)(b * H_ + h) << 6) + dh) * S_ + s] = bv;
    }
  }
}

// ---------------------------------------------------------------------------
// Kernel 3: wave-level flash attention; K tiles staged into LDS by the
// Tensor Data Mover (double-buffered, s_wait_tensorcnt), V via global b128.
// ---------------------------------------------------------------------------
#define AW 4   // waves per block
__global__ __launch_bounds__(128) void attn_kernel(
    const unsigned short* __restrict__ qp, const unsigned short* __restrict__ kp,
    const unsigned short* __restrict__ vT, const int* __restrict__ Qlen,
    const int* __restrict__ Vlen, float* __restrict__ out) {
  __shared__ __align__(16) unsigned short kbuf[AW][2][32 * KROW]; // TDM dest
  __shared__ __align__(16) unsigned short pbuf[AW][16 * 40];      // P transpose
  const int lane = threadIdx.x & 31;
  const int wave = threadIdx.x >> 5;
  const int tile = blockIdx.x * AW + wave;             // B*H*(S/16) = 8192
  const int b  = tile >> 10;
  const int r  = tile & 1023;
  const int h  = r >> 6;
  const int q0 = (r & 63) << 4;
  const int half = lane >> 4, lrow = lane & 15, koff = half * 8;

  const unsigned short* qbase = qp + (size_t)(b * H_ + h) * S_ * DH_;
  const unsigned short* kbase = kp + (size_t)(b * H_ + h) * S_ * DH_;
  const unsigned short* vbase = vT + (size_t)(b * H_ + h) * DH_ * S_;
  const int vlen = Vlen[b], qlen = Qlen[b];

  // Q tile 16x64 held in registers as two K=32 A-fragments
  v16bf aq[2];
  {
    const unsigned short* qrow = qbase + (size_t)(q0 + lrow) * DH_;
    #pragma unroll
    for (int c = 0; c < 2; ++c)
      aq[c] = cat8(*(const v8bf*)(qrow + c * 32 + koff),
                   *(const v8bf*)(qrow + c * 32 + koff + 16));
  }

  float mrow[8], lsum[8];
  #pragma unroll
  for (int i = 0; i < 8; ++i) { mrow[i] = -1.0e30f; lsum[i] = 0.0f; }
  v8f acc[4] = {};
  unsigned short* pb = &pbuf[wave][0];

  const int kmax = q0 + 16;                            // causal upper bound
  tdm_load_k(kbase, &kbuf[wave][0][0]);                // prologue tensor load
  for (int k0 = 0; k0 < kmax; k0 += 32) {
    const int cur = (k0 >> 5) & 1;
    const bool more = (k0 + 32) < kmax;
    if (more)                                          // next K tile in flight
      tdm_load_k(kbase + (size_t)(k0 + 32) * DH_, &kbuf[wave][cur ^ 1][0]);
    // V B-fragments (global) issued early: overlap tensor wait + softmax
    v16bf vb[4];
    #pragma unroll
    for (int nt = 0; nt < 4; ++nt) {
      const unsigned short* vr = vbase + (size_t)(nt * 16 + lrow) * S_ + k0 + koff;
      vb[nt] = cat8(*(const v8bf*)vr, *(const v8bf*)(vr + 16));
    }
    if (more) __builtin_amdgcn_s_wait_tensorcnt(1);    // oldest (cur) done
    else      __builtin_amdgcn_s_wait_tensorcnt(0);
    asm volatile("" ::: "memory");

    // --- S = q . k^T from LDS K tile (conflict-free: row stride 72 el) ---
    const unsigned short* kt = &kbuf[wave][cur][0];
    v8f st[2];
    #pragma unroll
    for (int t = 0; t < 2; ++t) {
      v8f s = {};
      const unsigned short* krow = kt + (t * 16 + lrow) * KROW;
      #pragma unroll
      for (int c = 0; c < 2; ++c) {
        v16bf bk = cat8(*(const v8bf*)(krow + c * 32 + koff),
                        *(const v8bf*)(krow + c * 32 + koff + 16));
        s = wmma_bf16(aq[c], bk, s);
      }
      st[t] = s;
    }
    // --- masks + online softmax (row i / i+8 across each 16-lane half) ---
    const int c0 = k0 + lrow, c1 = k0 + 16 + lrow;
    #pragma unroll
    for (int i = 0; i < 8; ++i) {
      const int qr = q0 + i + half * 8;
      float s0 = st[0][i] * 0.125f;                    // 1/sqrt(64)
      float s1 = st[1][i] * 0.125f;
      if (c0 >= vlen) s0 -= 1.0e12f;                   // key padding mask
      if (c0 > qr)    s0 -= 1.0e12f;                   // causal mask
      if (c1 >= vlen) s1 -= 1.0e12f;
      if (c1 > qr)    s1 -= 1.0e12f;
      float mt = fmaxf(s0, s1);
      mt = fmaxf(mt, __shfl_xor(mt, 1));
      mt = fmaxf(mt, __shfl_xor(mt, 2));
      mt = fmaxf(mt, __shfl_xor(mt, 4));
      mt = fmaxf(mt, __shfl_xor(mt, 8));
      const float mnew = fmaxf(mrow[i], mt);
      const float p0 = __expf(s0 - mnew);
      const float p1 = __expf(s1 - mnew);
      const float corr = __expf(mrow[i] - mnew);
      float ps = p0 + p1;
      ps += __shfl_xor(ps, 1);
      ps += __shfl_xor(ps, 2);
      ps += __shfl_xor(ps, 4);
      ps += __shfl_xor(ps, 8);
      lsum[i] = lsum[i] * corr + ps;
      mrow[i] = mnew;
      #pragma unroll
      for (int nt = 0; nt < 4; ++nt) acc[nt][i] *= corr;
      const int prow = (i + half * 8) * 40;            // C-layout -> LDS
      pb[prow + lrow]      = f2bf(p0);
      pb[prow + 16 + lrow] = f2bf(p1);
    }
    // --- read P back in A-layout (same-wave DS ops are in order) ---
    const unsigned short* pr = pb + lrow * 40 + koff;
    v16bf ap = cat8(*(const v8bf*)pr, *(const v8bf*)(pr + 16));
    // --- O += P(16x32) . v(32x64) ---
    #pragma unroll
    for (int nt = 0; nt < 4; ++nt) acc[nt] = wmma_bf16(ap, vb[nt], acc[nt]);
  }

  // --- normalize, apply query mask, store fp32 [B, S, H*Dh] ---
  #pragma unroll
  for (int i = 0; i < 8; ++i) {
    const int qr = q0 + i + half * 8;
    const float qm = (qr < qlen) ? (1.0f / lsum[i]) : 0.0f;
    float* orow = out + ((size_t)b * S_ + qr) * N_ + h * DH_;
    #pragma unroll
    for (int nt = 0; nt < 4; ++nt)
      orow[nt * 16 + lrow] = acc[nt][i] * qm;
  }
}

// ---------------------------------------------------------------------------
extern "C" void kernel_launch(void* const* d_in, const int* in_sizes, int n_in,
                              void* d_out, int out_size, void* d_ws, size_t ws_size,
                              hipStream_t stream) {
  (void)in_sizes; (void)n_in; (void)out_size; (void)ws_size;
  const float* Q  = (const float*)d_in[0];
  const float* K  = (const float*)d_in[1];
  const float* V  = (const float*)d_in[2];
  const float* WQ = (const float*)d_in[3];
  const float* WK = (const float*)d_in[4];
  const float* WV = (const float*)d_in[5];
  const int* Qlen = (const int*)d_in[6];
  const int* Vlen = (const int*)d_in[7];
  float* out = (float*)d_out;

  // Workspace (bf16 as ushort): Wt 3*1M, qp/kp/vT 8M each  => ~56.6 MB
  unsigned short* ws = (unsigned short*)d_ws;
  unsigned short* Wt = ws;
  unsigned short* qp = ws + (size_t)3 * 1024 * 1024;
  unsigned short* kp = qp + (size_t)8388608;
  unsigned short* vT = kp + (size_t)8388608;

  wconv_kernel<<<dim3(3072),  dim3(256), 0, stream>>>(WQ, WK, WV, Wt);
  proj_kernel <<<dim3(3072),  dim3(256), 0, stream>>>(Q, K, V, Wt, qp, kp, vT);
  attn_kernel <<<dim3(2048),  dim3(128), 0, stream>>>(qp, kp, vT, Qlen, Vlen, out);
}
